// Grid_61916248539356
// MI455X (gfx1250) — matrix operand order repaired and checked by reading, not compiled, not run on hardware
//
#include <hip/hip_runtime.h>
#include <math.h>

// ---------------------------------------------------------------------------
// Reference:
//   soft = softmax(indices[R,256]) @ codebook[256,16]      (WMMA f16 GEMM)
//   grid = (codebook[argmax(indices,1)] - soft) + soft     ([R,16] in d_ws)
//   out  = Catmull-Rom cubic interp of grid rows at N coords -> [N,16] f32
//
// MI455X plan:
//   Phase 1 is bandwidth-bound (256 MB indices @ 23.3 TB/s ~= 12 us): feed it
//   with the Tensor Data Mover (tensor_load_to_lds, TENSORcnt) and compute the
//   16x16x256 GEMM with 8x v_wmma_f32_16x16x32_f16 per wave.
//   Phase 2 is bandwidth-bound (256 MB output); the 16 MB grid table is
//   L2-resident (192 MB L2) so the 4 gathers/coord stay on-chip.
// ---------------------------------------------------------------------------

typedef __attribute__((ext_vector_type(16))) _Float16     v16h;
typedef __attribute__((ext_vector_type(8)))  float        v8f;
typedef __attribute__((ext_vector_type(4)))  float        v4f;
typedef __attribute__((ext_vector_type(4)))  unsigned int v4u;
typedef __attribute__((ext_vector_type(8)))  unsigned int v8u;

#define KDIM 256   // codebook entries (K of the GEMM)
#define CDIM 16    // channels (N of the GEMM)

// ---------------------------------------------------------------------------
// Kernel A: one wave (32 lanes) builds one 16-row tile of `grid`.
//   - 16x256 f32 indices tile pulled into LDS by one TDM tensor_load_to_lds
//   - codebook staged as f16 in LDS with plain loads (overlaps the TDM)
//   - softmax stats per row; row m is split across lanes m and m+16 with the
//     exact K-partitioning the WMMA A-fragment wants:
//       lane<16  owns K in {0..7, 16..23}  (+32*k0)
//       lane>=16 owns K in {8..15, 24..31} (+32*k0)
//   - 8x v_wmma_f32_16x16x32_f16 accumulate soft = P @ CB
// ---------------------------------------------------------------------------
__global__ __launch_bounds__(32) void grid_build_kernel(
    const float* __restrict__ codebook,   // [256,16] f32
    const float* __restrict__ indices,    // [R,256]  f32
    float* __restrict__ grid,             // [R,16]   f32 (workspace)
    int R)
{
  __shared__ float    s_idx[16 * KDIM];     // 16 KB (TDM destination)
  __shared__ _Float16 s_cb[KDIM * CDIM];    //  8 KB

  const int lane = (int)threadIdx.x;        // 0..31
  const int r0   = (int)blockIdx.x * 16;

  // ---- TDM: one tensor_load_to_lds DMAs the whole 16x256 f32 tile ---------
  // D# per CDNA5 ISA. 2D tensor: groups 2/3 are NULL.
  {
    unsigned int lds_base = (unsigned int)(uintptr_t)(&s_idx[0]);
    unsigned long long gaddr =
        (unsigned long long)(uintptr_t)indices +
        (unsigned long long)r0 * (KDIM * 4ull);

    v4u g0;
    g0[0] = 1u;                                   // count=1 valid user D#
    g0[1] = lds_base;                             // lds_addr (bytes)
    g0[2] = (unsigned int)(gaddr & 0xFFFFFFFFull);        // global_addr[31:0]
    g0[3] = (unsigned int)((gaddr >> 32) & 0x01FFFFFFull) // global_addr[56:32]
          | (2u << 30);                           // type=2 ("image")

    v8u g1;
    g1[0] = (2u << 16);                           // wg_mask=0, data_size=4B
    g1[1] = ((unsigned int)(KDIM & 0xFFFF)) << 16; // tensor_dim0[15:0]=256
    g1[2] = ((unsigned int)KDIM >> 16)            // tensor_dim0[31:16]
          | (((unsigned int)R & 0xFFFF) << 16);   // tensor_dim1[15:0]
    g1[3] = ((unsigned int)R >> 16)               // tensor_dim1[31:16]
          | ((unsigned int)KDIM << 16);           // tile_dim0=256
    g1[4] = 16u;                                  // tile_dim1=16, tile_dim2=0
    g1[5] = (unsigned int)KDIM;                   // tensor_dim0_stride[31:0]
    g1[6] = 0u;                                   // stride0[47:32], stride1 lo
    g1[7] = 0u;                                   // stride1 hi (unused, 2D)

    asm volatile("tensor_load_to_lds %0, %1"
                 :: "s"(g0), "s"(g1)
                 : "memory");
  }

  // ---- Stage codebook f32 -> f16 into LDS while the TDM runs --------------
  for (int i = 0; i < (KDIM * CDIM) / 32; ++i) {
    int idx = i * 32 + lane;
    s_cb[idx] = (_Float16)codebook[idx];
  }

  __builtin_amdgcn_s_wait_tensorcnt(0);   // TDM tile landed in LDS
  __syncthreads();

  const int row  = lane & 15;
  const int hsel = (lane >> 4) ? 8 : 0;     // which K-octets this lane owns
  const float* rp = s_idx + row * KDIM;

  // Pass 1: partial max + argmax over this lane's 128 K values.
  float mx = -3.4e38f; int amax = 0;
  for (int k0 = 0; k0 < KDIM; k0 += 32) {
    #pragma unroll
    for (int j = 0; j < 8; ++j) {
      int K1 = k0 + hsel + j;
      int K2 = k0 + 16 + hsel + j;
      float v1 = rp[K1], v2 = rp[K2];
      if (v1 > mx)                       { mx = v1; amax = K1; }
      else if (v1 == mx && K1 < amax)    { amax = K1; }
      if (v2 > mx)                       { mx = v2; amax = K2; }
      else if (v2 == mx && K2 < amax)    { amax = K2; }
    }
  }
  // Combine the two half-rows (lane L <-> lane L+16). wave32: width 32.
  {
    float mx_o  = __shfl_xor(mx, 16, 32);
    int   am_o  = __shfl_xor(amax, 16, 32);
    if (mx_o > mx)                      { amax = am_o; }
    else if (mx_o == mx && am_o < amax) { amax = am_o; }
    mx = fmaxf(mx, mx_o);               // row max, now uniform per row pair
  }

  // Pass 2: sum of exp(x - rowmax).
  float sum = 0.0f;
  for (int k0 = 0; k0 < KDIM; k0 += 32) {
    #pragma unroll
    for (int j = 0; j < 8; ++j) {
      sum += expf(rp[k0 + hsel + j]      - mx);
      sum += expf(rp[k0 + 16 + hsel + j] - mx);
    }
  }
  sum += __shfl_xor(sum, 16, 32);
  const float inv = 1.0f / sum;

  // WMMA: soft[16x16] += P[16x32] * CB[32x16], K = 256 -> 8 fully unrolled.
  // A frag (16-bit A 16x32, ISA 7.12.2): lane<16 row=lane, halfs 0..7 = K k0+0..7,
  //   halfs 8..15 = K k0+16..23; lane>=16 gets the +8 octets.
  // B frag (32x16): lanes 0..15 col=lane hold K=k0+0..7, k0+16..23,
  //   lanes 16..31 col=lane-16 hold K=k0+8..15, k0+24..31 (same as A).
  v8f acc = {};
  const int col    = lane & 15;
  const int kb_off = (lane >> 4) ? 8 : 0;
  #pragma unroll
  for (int k0 = 0; k0 < KDIM; k0 += 32) {
    v16h a, b;
    #pragma unroll
    for (int j = 0; j < 8; ++j) {
      a[j]     = (_Float16)(expf(rp[k0 + hsel + j]      - mx) * inv);
      a[j + 8] = (_Float16)(expf(rp[k0 + 16 + hsel + j] - mx) * inv);
      b[j]     = s_cb[(k0 + kb_off + j)     * CDIM + col];
      b[j + 8] = s_cb[(k0 + 16 + kb_off + j) * CDIM + col];
    }
    // 8 args: (neg_a, A, neg_b, B, c_mod, C, reuse_a, reuse_b)
    acc = __builtin_amdgcn_wmma_f32_16x16x32_f16(
        false, a, false, b, (short)0, acc, false, false);
  }

  // C/D layout: VGPR g, lane<16 -> M=g, N=lane; lane>=16 -> M=g+8, N=lane-16.
  // grid = (max_grid - soft) + soft  (straight-through; f32 codebook read)
  #pragma unroll
  for (int g = 0; g < 8; ++g) {
    int m  = g + ((lane >> 4) ? 8 : 0);
    int am = __shfl(amax, m, 32);            // row m's argmax lives in lane m
    float soft = acc[g];
    float mg   = codebook[am * CDIM + col];
    grid[(size_t)(r0 + m) * CDIM + col] = (mg - soft) + soft;
  }
}

// ---------------------------------------------------------------------------
// Kernel B: cubic interpolation. 4 threads per coordinate; thread (n, q)
// produces channels 4q..4q+3 so stores across a wave are fully coalesced.
// ---------------------------------------------------------------------------
__global__ __launch_bounds__(256) void interp_kernel(
    const float* __restrict__ coords,   // [N]
    const float* __restrict__ grid,     // [R,16]
    float* __restrict__ out,            // [N,16]
    int N, int R)
{
  int t = (int)blockIdx.x * 256 + (int)threadIdx.x;
  int n = t >> 2;
  int q = t & 3;
  if (n >= N) return;

  // Pull upcoming coords toward the caches (gfx1250 global_prefetch_b8).
  __builtin_prefetch(coords + n + 4096, 0, 1);

  float x = coords[n];
  float c = (x + 1.0f) * 0.5f * (float)(R - 1);

  float fl = floorf(c);
  int left   = (int)fminf(fl, (float)(R - 2));
  int right  = (int)fmaxf(ceilf(c), 1.0f);
  int lleft  = (int)fminf(fmaxf(floorf(c - 1.0f), 0.0f), (float)(R - 3));
  int rright = (int)fminf(fmaxf(ceilf(c + 1.0f), 2.0f), (float)(R - 1));
  float w = c - (float)left;

  const v4f* gp = (const v4f*)grid;        // [R][4] of float4
  v4f lv  = gp[left   * 4 + q];
  v4f rv  = gp[right  * 4 + q];
  v4f llv = gp[lleft  * 4 + q];
  v4f rrv = gp[rright * 4 + q];

  v4f res = lv + 0.5f * w * (
      rv - llv + w * (2.0f * llv - 5.0f * lv + 4.0f * rv - rrv
                      + w * (3.0f * (lv - rv) + rrv - llv)));

  ((v4f*)out)[(size_t)n * 4 + q] = res;
}

// ---------------------------------------------------------------------------
extern "C" void kernel_launch(void* const* d_in, const int* in_sizes, int n_in,
                              void* d_out, int out_size, void* d_ws, size_t ws_size,
                              hipStream_t stream) {
  const float* coords   = (const float*)d_in[0];   // [N]
  const float* codebook = (const float*)d_in[1];   // [256,16]
  const float* indices  = (const float*)d_in[2];   // [R,256]
  float* out = (float*)d_out;

  const int N = in_sizes[0];
  const int R = in_sizes[2] / KDIM;

  float* grid_ws = (float*)d_ws;                   // R*16 f32 = 16 MB scratch

  // Phase 1: build grid[R,16] with TDM staging + softmax + WMMA GEMM.
  grid_build_kernel<<<R / 16, 32, 0, stream>>>(codebook, indices, grid_ws, R);

  // Phase 2: cubic interpolation at N coords, 4 threads per coord.
  long long threads = (long long)N * 4;
  int blocks = (int)((threads + 255) / 256);
  interp_kernel<<<blocks, 256, 0, stream>>>(coords, grid_ws, out, N, R);
}
